// CustomGraphNet_40681930227733
// MI455X (gfx1250) — matrix operand order repaired and checked by reading, not compiled, not run on hardware
//
#include <hip/hip_runtime.h>

typedef __attribute__((ext_vector_type(16))) __bf16 v16bf;
typedef __attribute__((ext_vector_type(8)))  __bf16 v8bf;
typedef __attribute__((ext_vector_type(4)))  __bf16 v4bf;
typedef __attribute__((ext_vector_type(8)))  float  v8f;
typedef __attribute__((ext_vector_type(4)))  float  v4f;

#define LDIM 128

// ---------------------------------------------------------------------------
// Weight swizzle: fp32 [Din x 128] -> bf16 pre-swizzled WMMA B-fragment layout.
// Layout: [kt][ct][lane][e] ; kt = K-tile (32), ct = col-tile (16), lane 0..31,
// e 0..15 with  K = kt*32 + (lane>>4)*16 + e,  N = ct*16 + (lane&15).
// Each lane's 16 bf16 (32B) are contiguous -> 2x global_load_b128 per fragment.
// ---------------------------------------------------------------------------
__global__ void swz_bf16_kernel(const float* __restrict__ W, __bf16* __restrict__ o,
                                int Din, int nkt) {
    int i = blockIdx.x * blockDim.x + threadIdx.x;
    if (i >= nkt * 4096) return;
    int e    = i & 15;
    int lane = (i >> 4) & 31;
    int ct   = (i >> 9) & 7;
    int kt   = i >> 12;
    int kk = (kt << 5) + ((lane >> 4) << 4) + e;
    int n  = (ct << 4) + (lane & 15);
    o[i] = (kk < Din) ? (__bf16)W[(size_t)kk * LDIM + n] : (__bf16)0.0f;
}

__global__ void zero_f32_kernel(float* __restrict__ p, int n) {
    int i = blockIdx.x * blockDim.x + threadIdx.x;
    if (i < n) p[i] = 0.0f;
}

// ---------------------------------------------------------------------------
// A-fragment from a row-major bf16 LDS tile (stride in elements, 16B aligned):
// lane's 16 values = two contiguous 8-element runs.
// ---------------------------------------------------------------------------
__device__ __forceinline__ v16bf load_afrag(const __bf16* rowp, int k0, int kh) {
    v8bf a0 = *(const v8bf*)(rowp + k0 + 2 * kh);
    v8bf a1 = *(const v8bf*)(rowp + k0 + 2 * kh + 8);
    return __builtin_shufflevector(a0, a1, 0,1,2,3,4,5,6,7,8,9,10,11,12,13,14,15);
}

__device__ __forceinline__ v8f wmma_bf16(v16bf a, v16bf b, v8f c) {
    return __builtin_amdgcn_wmma_f32_16x16x32_bf16(false, a, false, b, (short)0, c,
                                                   false, false);
}

// ---------------------------------------------------------------------------
// Fused ForwardNet with LayerNorm (+optional residual, +gather/concat).
//   out = LN( relu(X @ W1 + b1) @ W2 + b2 ) [+ res]
// 256 threads = 8 wave32; block computes 16 rows x 128 cols; wave w owns cols
// [16w,16w+16). W1sw/W2sw are pre-swizzled B-fragment weights.
// ---------------------------------------------------------------------------
__global__ __launch_bounds__(256)
void fnet_ln_kernel(const float* __restrict__ X0, const float* __restrict__ X1,
                    const float* __restrict__ X2,
                    const int* __restrict__ idx0, const int* __restrict__ idx1,
                    int d0, int d1, int d2, int Din, int R,
                    const __bf16* __restrict__ W1sw, const float* __restrict__ b1,
                    const __bf16* __restrict__ W2sw, const float* __restrict__ b2,
                    const float* __restrict__ g, const float* __restrict__ be,
                    const float* __restrict__ res, float* __restrict__ Out)
{
    __shared__ __align__(16) __bf16 As[16][392];   // input rows (K up to 384)
    __shared__ __align__(16) __bf16 Ah[16][136];   // hidden activations
    __shared__ float Acc[16][LDIM];                // pre-LN output
    __shared__ float psum[16][17], psq[16][17];
    __shared__ float red[16][2];

    const int tid  = threadIdx.x;
    const int lane = tid & 31;
    const int wv   = tid >> 5;
    const int blockRow = blockIdx.x * 16;
    const int Dinp = (Din + 31) & ~31;

    // ---- gather/concat 16 rows into As (bf16), float4-vectorized ----
    {
        const int nq = Dinp >> 2;          // 4-element groups per row
        for (int t = tid; t < 16 * nq; t += 256) {
            int r = t / nq, c4 = (t - r * nq) << 2;
            int rg = blockRow + r; if (rg >= R) rg = R - 1;
            v4f f = {0.0f, 0.0f, 0.0f, 0.0f};
            if (c4 < Din) {
                if (c4 < d0) {
                    int row = idx0 ? idx0[rg] : rg;
                    f = *(const v4f*)(X0 + (size_t)row * d0 + c4);
                } else if (c4 < d0 + d1) {
                    int row = idx1 ? idx1[rg] : rg;
                    f = *(const v4f*)(X1 + (size_t)row * d1 + (c4 - d0));
                } else {
                    f = *(const v4f*)(X2 + (size_t)rg * d2 + (c4 - d0 - d1));
                }
            }
            *(v4bf*)&As[r][c4] = __builtin_convertvector(f, v4bf);
        }
    }
    __syncthreads();

    const int m    = lane & 15;          // A row
    const int kh   = (lane >> 4) * 8;    // A K-half offset
    const int colg = wv * 16 + (lane & 15);

    // ---- GEMM1: [16 x Din] @ [Din x 128], B fragments straight from L2 ----
    v8f acc = {};
    const int nkt1 = Dinp >> 5;
    for (int kt = 0; kt < nkt1; ++kt) {
        v16bf a = load_afrag(&As[m][0], kt << 5, kh);
        v16bf b = *(const v16bf*)(W1sw + ((((size_t)kt << 3) + wv) * 32 + lane) * 16);
        acc = wmma_bf16(a, b, acc);
    }

    // ---- bias + ReLU -> bf16 hidden ----
    {
        float bias1 = b1[colg];
#pragma unroll
        for (int i = 0; i < 8; ++i) {
            int mrow = i + ((lane >> 4) << 3);
            Ah[mrow][colg] = (__bf16)fmaxf(acc[i] + bias1, 0.0f);
        }
    }
    __syncthreads();

    // ---- GEMM2: [16 x 128] @ [128 x 128] ----
    v8f acc2 = {};
#pragma unroll
    for (int kt = 0; kt < 4; ++kt) {
        v16bf a = load_afrag(&Ah[m][0], kt << 5, kh);
        v16bf b = *(const v16bf*)(W2sw + ((((size_t)kt << 3) + wv) * 32 + lane) * 16);
        acc2 = wmma_bf16(a, b, acc2);
    }

    // ---- bias -> Acc ----
    {
        float bias2 = b2[colg];
#pragma unroll
        for (int i = 0; i < 8; ++i) {
            int mrow = i + ((lane >> 4) << 3);
            Acc[mrow][colg] = acc2[i] + bias2;
        }
    }
    __syncthreads();

    // ---- LayerNorm stats (two-level parallel reduction) ----
    {
        int r = tid >> 4, t = tid & 15;
        float s = 0.0f, s2 = 0.0f;
#pragma unroll
        for (int j = 0; j < 8; ++j) {
            float v = Acc[r][t * 8 + j];
            s += v; s2 += v * v;
        }
        psum[r][t] = s; psq[r][t] = s2;
    }
    __syncthreads();
    if (tid < 16) {
        float su = 0.0f, sq = 0.0f;
#pragma unroll
        for (int j = 0; j < 16; ++j) { su += psum[tid][j]; sq += psq[tid][j]; }
        float mu  = su * (1.0f / LDIM);
        float var = sq * (1.0f / LDIM) - mu * mu;
        red[tid][0] = mu;
        red[tid][1] = rsqrtf(var + 1e-5f);
    }
    __syncthreads();

    // ---- apply LN (+residual) and store ----
    for (int t = tid; t < 16 * LDIM; t += 256) {
        int r = t >> 7, c = t & 127;
        int rg = blockRow + r;
        if (rg >= R) continue;
        float v = g[c] * (Acc[r][c] - red[r][0]) * red[r][1] + be[c];
        if (res) v += res[(size_t)rg * LDIM + c];
        Out[(size_t)rg * LDIM + c] = v;
    }
}

// ---------------------------------------------------------------------------
// Attention score: s = relu(edge @ attW + b); per-destination atomic max.
// Scores >= 0, so int-bitwise atomicMax is order-correct vs init 0.
// ---------------------------------------------------------------------------
__global__ __launch_bounds__(256)
void attn_score_kernel(const float* __restrict__ edge, const float* __restrict__ attW,
                       const float* __restrict__ attb, const int* __restrict__ dstIdx,
                       float* __restrict__ score, int* __restrict__ segmax, int E)
{
    __shared__ v4f w4[32];
    int t = threadIdx.x;
    if (t < 32) w4[t] = *(const v4f*)(attW + t * 4);
    __syncthreads();
    int e = blockIdx.x * blockDim.x + t;
    if (e >= E) return;
    const v4f* row = (const v4f*)(edge + (size_t)e * LDIM);
    float s = attb[0];
#pragma unroll
    for (int k = 0; k < 32; ++k) {
        v4f f = row[k], w = w4[k];
        s += f[0] * w[0] + f[1] * w[1] + f[2] * w[2] + f[3] * w[3];
    }
    s = fmaxf(s, 0.0f);
    score[e] = s;
    atomicMax(&segmax[dstIdx[e]], __float_as_int(s));
}

__global__ __launch_bounds__(256)
void attn_exp_kernel(const float* __restrict__ score, const int* __restrict__ dstIdx,
                     const int* __restrict__ segmax, float* __restrict__ wexp,
                     float* __restrict__ segsum, int E)
{
    int e = blockIdx.x * blockDim.x + threadIdx.x;
    if (e >= E) return;
    int d = dstIdx[e];
    float v = __expf(score[e] - __int_as_float(segmax[d]));
    wexp[e] = v;
    atomicAdd(&segsum[d], v);
}

// 32 threads per edge, 4 columns each: agg[dst] += alpha * edge
__global__ __launch_bounds__(256)
void attn_agg_kernel(const float* __restrict__ edge, const float* __restrict__ wexp,
                     const float* __restrict__ segsum, const int* __restrict__ dstIdx,
                     float* __restrict__ agg, int E)
{
    long long tid = (long long)blockIdx.x * blockDim.x + threadIdx.x;
    int e = (int)(tid >> 5);
    int part = (int)(tid & 31);
    if (e >= E) return;
    int d = dstIdx[e];
    float alpha = wexp[e] / segsum[d];
    const float* row = edge + (size_t)e * LDIM + part * 4;
    float* out = agg + (size_t)d * LDIM + part * 4;
#pragma unroll
    for (int j = 0; j < 4; ++j) atomicAdd(&out[j], alpha * row[j]);
}

// ---------------------------------------------------------------------------
// Decoder: relu(node @ W1 + b1) [WMMA] then 128->3 scalar tail (no LN)
// ---------------------------------------------------------------------------
__global__ __launch_bounds__(256)
void decoder_kernel(const float* __restrict__ node,
                    const __bf16* __restrict__ W1sw, const float* __restrict__ b1,
                    const float* __restrict__ W2, const float* __restrict__ b2,
                    float* __restrict__ Out, int R, int Dout)
{
    __shared__ __align__(16) __bf16 As[16][136];
    __shared__ float Acc[16][LDIM];

    const int tid  = threadIdx.x;
    const int lane = tid & 31;
    const int wv   = tid >> 5;
    const int blockRow = blockIdx.x * 16;

    for (int t = tid; t < 16 * 32; t += 256) {
        int r = t >> 5, c4 = (t & 31) << 2;
        int rg = blockRow + r; if (rg >= R) rg = R - 1;
        v4f f = *(const v4f*)(node + (size_t)rg * LDIM + c4);
        *(v4bf*)&As[r][c4] = __builtin_convertvector(f, v4bf);
    }
    __syncthreads();

    const int m    = lane & 15;
    const int kh   = (lane >> 4) * 8;
    const int colg = wv * 16 + (lane & 15);

    v8f acc = {};
#pragma unroll
    for (int kt = 0; kt < 4; ++kt) {
        v16bf a = load_afrag(&As[m][0], kt << 5, kh);
        v16bf b = *(const v16bf*)(W1sw + ((((size_t)kt << 3) + wv) * 32 + lane) * 16);
        acc = wmma_bf16(a, b, acc);
    }
    {
        float bias1 = b1[colg];
#pragma unroll
        for (int i = 0; i < 8; ++i) {
            int mrow = i + ((lane >> 4) << 3);
            Acc[mrow][colg] = fmaxf(acc[i] + bias1, 0.0f);
        }
    }
    __syncthreads();

    if (tid < 16 * Dout) {
        int r = tid / Dout, j = tid - r * Dout;
        int rg = blockRow + r;
        if (rg < R) {
            float s = b2[j];
            for (int k = 0; k < LDIM; ++k) s += Acc[r][k] * W2[k * Dout + j];
            Out[(size_t)rg * Dout + j] = s;
        }
    }
}

// ---------------------------------------------------------------------------
// Host launch
// ---------------------------------------------------------------------------
extern "C" void kernel_launch(void* const* d_in, const int* in_sizes, int n_in,
                              void* d_out, int out_size, void* d_ws, size_t ws_size,
                              hipStream_t stream)
{
    const float* x         = (const float*)d_in[0];
    const float* edge_attr = (const float*)d_in[1];
    const int*   eidx      = (const int*)d_in[2];
    const float* en_W1 = (const float*)d_in[3];  const float* en_b1 = (const float*)d_in[4];
    const float* en_W2 = (const float*)d_in[5];  const float* en_b2 = (const float*)d_in[6];
    const float* en_g  = (const float*)d_in[7];  const float* en_be = (const float*)d_in[8];
    const float* ee_W1 = (const float*)d_in[9];  const float* ee_b1 = (const float*)d_in[10];
    const float* ee_W2 = (const float*)d_in[11]; const float* ee_b2 = (const float*)d_in[12];
    const float* ee_g  = (const float*)d_in[13]; const float* ee_be = (const float*)d_in[14];
    const float* pe_W1 = (const float*)d_in[15]; const float* pe_b1 = (const float*)d_in[16];
    const float* pe_W2 = (const float*)d_in[17]; const float* pe_b2 = (const float*)d_in[18];
    const float* pe_g  = (const float*)d_in[19]; const float* pe_be = (const float*)d_in[20];
    const float* pn_W1 = (const float*)d_in[21]; const float* pn_b1 = (const float*)d_in[22];
    const float* pn_W2 = (const float*)d_in[23]; const float* pn_b2 = (const float*)d_in[24];
    const float* pn_g  = (const float*)d_in[25]; const float* pn_be = (const float*)d_in[26];
    const float* att_W = (const float*)d_in[27]; const float* att_b = (const float*)d_in[28];
    const float* dd_W1 = (const float*)d_in[29]; const float* dd_b1 = (const float*)d_in[30];
    const float* dd_W2 = (const float*)d_in[31]; const float* dd_b2 = (const float*)d_in[32];

    const int N = in_sizes[0] / 16;
    const int E = in_sizes[1] / 8;
    const int P = in_sizes[15] / (3 * LDIM * LDIM);
    const int L = LDIM;

    const int* srcIdx = eidx;       // edge_index[0] = x_j (source)
    const int* dstIdx = eidx + E;   // edge_index[1] = x_i (target)

    size_t off = 0;
    auto alloc = [&](size_t bytes) -> void* {
        void* p = (char*)d_ws + off;
        off += (bytes + 255) & ~(size_t)255;
        return p;
    };
    float*  node   = (float*)alloc((size_t)N * L * 4);
    float*  edge   = (float*)alloc((size_t)E * L * 4);
    float*  agg    = (float*)alloc((size_t)N * L * 4);
    float*  segsum = (float*)alloc((size_t)N * 4);
    int*    segmax = (int*)  alloc((size_t)N * 4);
    float*  score  = (float*)alloc((size_t)E * 4);
    float*  wexp   = (float*)alloc((size_t)E * 4);

    // swizzled weight buffers: nkt * 4096 bf16 elements each
    const size_t SW1  = 1 * 4096;    // Din=16 or 8  -> 1 k-tile
    const size_t SW4  = 4 * 4096;    // Din=128      -> 4 k-tiles
    const size_t SW8  = 8 * 4096;    // Din=256      -> 8 k-tiles
    const size_t SW12 = 12 * 4096;   // Din=384      -> 12 k-tiles
    __bf16* en_W1s = (__bf16*)alloc(SW1 * 2);
    __bf16* en_W2s = (__bf16*)alloc(SW4 * 2);
    __bf16* ee_W1s = (__bf16*)alloc(SW1 * 2);
    __bf16* ee_W2s = (__bf16*)alloc(SW4 * 2);
    __bf16* pe_W1s = (__bf16*)alloc((size_t)P * SW12 * 2);
    __bf16* pe_W2s = (__bf16*)alloc((size_t)P * SW4 * 2);
    __bf16* pn_W1s = (__bf16*)alloc((size_t)P * SW8 * 2);
    __bf16* pn_W2s = (__bf16*)alloc((size_t)P * SW4 * 2);
    __bf16* dd_W1s = (__bf16*)alloc(SW4 * 2);

    auto swz = [&](const float* W, __bf16* o, int Din, int nkt) {
        int tot = nkt * 4096;
        swz_bf16_kernel<<<(unsigned)((tot + 255) / 256), 256, 0, stream>>>(W, o, Din, nkt);
    };
    swz(en_W1, en_W1s, 16, 1);
    swz(en_W2, en_W2s, 128, 4);
    swz(ee_W1, ee_W1s, 8, 1);
    swz(ee_W2, ee_W2s, 128, 4);
    for (int p = 0; p < P; ++p) {
        swz(pe_W1 + (size_t)p * 3 * L * L, pe_W1s + (size_t)p * SW12, 384, 12);
        swz(pe_W2 + (size_t)p * L * L,     pe_W2s + (size_t)p * SW4,  128, 4);
        swz(pn_W1 + (size_t)p * 2 * L * L, pn_W1s + (size_t)p * SW8,  256, 8);
        swz(pn_W2 + (size_t)p * L * L,     pn_W2s + (size_t)p * SW4,  128, 4);
    }
    swz(dd_W1, dd_W1s, 128, 4);

    const unsigned nbN = (unsigned)((N + 15) / 16);
    const unsigned nbE = (unsigned)((E + 15) / 16);

    // Encoders
    fnet_ln_kernel<<<nbN, 256, 0, stream>>>(x, nullptr, nullptr, nullptr, nullptr,
        16, 0, 0, 16, N, en_W1s, en_b1, en_W2s, en_b2, en_g, en_be, nullptr, node);
    fnet_ln_kernel<<<nbE, 256, 0, stream>>>(edge_attr, nullptr, nullptr, nullptr, nullptr,
        8, 0, 0, 8, E, ee_W1s, ee_b1, ee_W2s, ee_b2, ee_g, ee_be, nullptr, edge);

    // Processor layers
    for (int p = 0; p < P; ++p) {
        // edge update: edge += fnet(concat(node[dst], node[src], edge))
        fnet_ln_kernel<<<nbE, 256, 0, stream>>>(node, node, edge, dstIdx, srcIdx,
            L, L, L, 3 * L, E,
            pe_W1s + (size_t)p * SW12, pe_b1 + (size_t)p * L,
            pe_W2s + (size_t)p * SW4,  pe_b2 + (size_t)p * L,
            pe_g + (size_t)p * L, pe_be + (size_t)p * L, edge, edge);

        zero_f32_kernel<<<(unsigned)(((size_t)N * L + 255) / 256), 256, 0, stream>>>(agg, N * L);
        zero_f32_kernel<<<(unsigned)((N + 255) / 256), 256, 0, stream>>>(segsum, N);
        zero_f32_kernel<<<(unsigned)((N + 255) / 256), 256, 0, stream>>>((float*)segmax, N);

        attn_score_kernel<<<(unsigned)((E + 255) / 256), 256, 0, stream>>>(
            edge, att_W + (size_t)p * L, att_b + p, dstIdx, score, segmax, E);
        attn_exp_kernel<<<(unsigned)((E + 255) / 256), 256, 0, stream>>>(
            score, dstIdx, segmax, wexp, segsum, E);
        attn_agg_kernel<<<(unsigned)(((size_t)E * 32 + 255) / 256), 256, 0, stream>>>(
            edge, wexp, segsum, dstIdx, agg, E);

        // node update: node += fnet(concat(node, agg))
        fnet_ln_kernel<<<nbN, 256, 0, stream>>>(node, agg, nullptr, nullptr, nullptr,
            L, L, 0, 2 * L, N,
            pn_W1s + (size_t)p * SW8, pn_b1 + (size_t)p * L,
            pn_W2s + (size_t)p * SW4, pn_b2 + (size_t)p * L,
            pn_g + (size_t)p * L, pn_be + (size_t)p * L, node, node);
    }

    // Decoder -> d_out [N, 3]
    decoder_kernel<<<nbN, 256, 0, stream>>>(node, dd_W1s, dd_b1, dd_W2, dd_b2,
                                            (float*)d_out, N, 3);
    (void)ws_size; (void)n_in; (void)out_size;
}
